// gNet_68341519614738
// MI455X (gfx1250) — matrix operand order repaired and checked by reading, not compiled, not run on hardware
//
#include <hip/hip_runtime.h>
#include <hip/hip_bf16.h>
#include <stdint.h>

typedef __attribute__((ext_vector_type(16))) __bf16 v16bf;
typedef __attribute__((ext_vector_type(8)))  __bf16 v8bf;
typedef __attribute__((ext_vector_type(8)))  float  v8f;

#define H_STEP 0.1f
#define LN_EPS 1e-5f

// ---------- bf16 helpers (RNE) ----------
__device__ __forceinline__ __bf16 f2bf(float f) {
  union { float f; uint32_t u; } a; a.f = f;
  uint32_t r = a.u + 0x7fffu + ((a.u >> 16) & 1u);
  union { uint16_t s; __bf16 b; } c; c.s = (uint16_t)(r >> 16);
  return c.b;
}
__device__ __forceinline__ float bf2f(__bf16 b) {
  union { uint16_t s; __bf16 b; } c; c.b = b;
  union { uint32_t u; float f; } a; a.u = ((uint32_t)c.s) << 16;
  return a.f;
}
__device__ __forceinline__ v16bf ld16bf(const __bf16* p0, const __bf16* p1) {
  v8bf x = *(const v8bf*)p0;
  v8bf y = *(const v8bf*)p1;
  v16bf r;
#pragma unroll
  for (int i = 0; i < 8; ++i) { r[i] = x[i]; r[8 + i] = y[i]; }
  return r;
}

// ---------- WMMA 16x16 tile over K, bf16 inputs / f32 accum ----------
// W: row-major [16][CDIM] (this wave's 16 output rows)
// tile: LDS, item-major [16][cpad] bf16 (tile[m][c] = X[c][m])
template<int CDIM>
__device__ __forceinline__ v8f wmma_rowtile(const __bf16* __restrict__ W,
                                            const __bf16* tile, int cpad,
                                            int lane, v8f acc) {
  const int mrow = lane & 15;   // A: M row / B: N col
  const int kh   = lane >> 4;   // lane-half selects K sub-ranges
#pragma unroll
  for (int kb = 0; kb < CDIM; kb += 32) {
    // A fragment: slots 0..7 -> K = kb + kh*8 + i ; slots 8..15 -> +16
    const __bf16* wp = W + (size_t)mrow * CDIM + kb + kh * 8;
    v16bf a = ld16bf(wp, wp + 16);
    // B fragment: slots 0..7 -> K = kb + kh*8 + i ; slots 8..15 -> +16, N = mrow
    const __bf16* bp = tile + mrow * cpad + kb + kh * 8;
    v16bf b = ld16bf(bp, bp + 16);
    acc = __builtin_amdgcn_wmma_f32_16x16x32_bf16(false, a, false, b,
                                                  (short)0, acc, false, false);
  }
  return acc;
}

// ---------- block sum/sumsq reduction -> 2 global atomics ----------
__device__ __forceinline__ void stats_reduce(float s, float q,
                                             float* redS, float* redQ,
                                             float* gstats) {
  const int lane = threadIdx.x & 31, w = threadIdx.x >> 5;
  const int nw = blockDim.x >> 5;
#pragma unroll
  for (int off = 16; off > 0; off >>= 1) {
    s += __shfl_down(s, off, 32);
    q += __shfl_down(q, off, 32);
  }
  if (lane == 0) { redS[w] = s; redQ[w] = q; }
  __syncthreads();
  if (w == 0) {
    float s2 = (lane < nw) ? redS[lane] : 0.f;
    float q2 = (lane < nw) ? redQ[lane] : 0.f;
#pragma unroll
    for (int off = 16; off > 0; off >>= 1) {
      s2 += __shfl_down(s2, off, 32);
      q2 += __shfl_down(q2, off, 32);
    }
    if (lane == 0) { atomicAdd(&gstats[0], s2); atomicAdd(&gstats[1], q2); }
  }
}

// ---------- weight prep ----------
__global__ void k_cvt(__bf16* dst, const float* src, int n) {
  int i = blockIdx.x * blockDim.x + threadIdx.x;
  if (i < n) dst[i] = f2bf(src[i]);
}
// KE1 [256][384] -> [256][256]: cols<128: A+B ; else: C
__global__ void k_fold_e(__bf16* dst, const float* src) {
  int i = blockIdx.x * blockDim.x + threadIdx.x;
  if (i >= 256 * 256) return;
  int o = i >> 8, c = i & 255;
  float v = (c < 128) ? (src[o * 384 + c] + src[o * 384 + 128 + c])
                      : src[o * 384 + 128 + c];
  dst[i] = f2bf(v);
}
// KN1 [256][384] -> [0.5A+B | 0.5A-B | C] acting on [s_i; s_j; xn]
__global__ void k_fold_n(__bf16* dst, const float* src) {
  int i = blockIdx.x * blockDim.x + threadIdx.x;
  if (i >= 256 * 384) return;
  int o = i / 384, c = i % 384;
  float v;
  if (c < 128)      v = 0.5f * src[o * 384 + c] + src[o * 384 + 128 + c];
  else if (c < 256) v = 0.5f * src[o * 384 + (c - 128)] - src[o * 384 + c];
  else              v = src[o * 384 + c];
  dst[i] = f2bf(v);
}

__global__ void k_finalize(float* slot, float inv) {
  if (threadIdx.x == 0 && blockIdx.x == 0) {
    float m = slot[0] * inv;
    float var = slot[1] * inv - m * m;
    slot[2] = m;
    slot[3] = rsqrtf(var + LN_EPS);
  }
}

// ---------- GEMM1 from f32 channel-major source (openers) ----------
// X: [CDIM, M] f32 ; T out: [M, ODIM] bf16 item-major ; stats: sum/sumsq
template<int CDIM, int ODIM>
__global__ void k_gemm1_f32(const float* __restrict__ X, const __bf16* __restrict__ W,
                            __bf16* __restrict__ T, float* __restrict__ stats, int M) {
  constexpr int CPAD = CDIM + 8;
  __shared__ __bf16 tile[16 * CPAD];
  __shared__ float redS[32], redQ[32];
  const int tid = threadIdx.x, lane = tid & 31, w = tid >> 5;
  const int m0 = blockIdx.x * 16;
  for (int idx = tid; idx < 16 * CDIM; idx += blockDim.x) {
    int c = idx >> 4, m = idx & 15;
    tile[m * CPAD + c] = f2bf(X[(size_t)c * M + m0 + m]);
  }
  __syncthreads();
  v8f acc = {};
  acc = wmma_rowtile<CDIM>(W + (size_t)(16 * w) * CDIM, tile, CPAD, lane, acc);
  const int m = m0 + (lane & 15);
  const int ob = 16 * w + (lane >> 4) * 8;
  v8bf pk; float s = 0.f, q = 0.f;
#pragma unroll
  for (int v = 0; v < 8; ++v) { float d = acc[v]; s += d; q += d * d; pk[v] = f2bf(d); }
  *(v8bf*)&T[(size_t)m * ODIM + ob] = pk;
  stats_reduce(s, q, redS, redQ, stats);
}

// ---------- edge GEMM1: gather concat [xn[iInd]; xe] (folded C=256) ----------
__global__ void k_egemm1(const __bf16* __restrict__ xnbf, const __bf16* __restrict__ xebf,
                         const int* __restrict__ iInd, const __bf16* __restrict__ W,
                         __bf16* __restrict__ T, float* __restrict__ stats, int E) {
  constexpr int CDIM = 256, CPAD = CDIM + 8, ODIM = 256;
  __shared__ __bf16 tile[16 * CPAD];
  __shared__ float redS[32], redQ[32];
  const int tid = threadIdx.x, lane = tid & 31, w = tid >> 5;
  const int e0 = blockIdx.x * 16;
  for (int idx = tid; idx < 16 * 32; idx += blockDim.x) {
    int m = idx >> 5, j = idx & 31;
    int e = e0 + m;
    const __bf16* src = (j < 16) ? (xnbf + (size_t)iInd[e] * 128 + j * 8)
                                 : (xebf + (size_t)e * 128 + (j - 16) * 8);
    *(uint4*)&tile[m * CPAD + j * 8] = *(const uint4*)src;
  }
  __syncthreads();
  v8f acc = {};
  acc = wmma_rowtile<CDIM>(W + (size_t)(16 * w) * CDIM, tile, CPAD, lane, acc);
  const int m = e0 + (lane & 15);
  const int ob = 16 * w + (lane >> 4) * 8;
  v8bf pk; float s = 0.f, q = 0.f;
#pragma unroll
  for (int v = 0; v < 8; ++v) { float d = acc[v]; s += d; q += d * d; pk[v] = f2bf(d); }
  *(v8bf*)&T[(size_t)m * ODIM + ob] = pk;
  stats_reduce(s, q, redS, redQ, stats);
}

// ---------- node GEMM1: [s_i; s_j; xn] (folded weights, C=384) ----------
__global__ void k_ngemm1(const float* __restrict__ si, const float* __restrict__ sj,
                         const __bf16* __restrict__ xnbf, const __bf16* __restrict__ W,
                         __bf16* __restrict__ T, float* __restrict__ stats, int N) {
  constexpr int CDIM = 384, CPAD = CDIM + 8, ODIM = 256;
  __shared__ __bf16 tile[16 * CPAD];
  __shared__ float redS[32], redQ[32];
  const int tid = threadIdx.x, lane = tid & 31, w = tid >> 5;
  const int n0 = blockIdx.x * 16;
  for (int idx = tid; idx < 16 * 256; idx += blockDim.x) {
    int c = idx >> 4, m = idx & 15;
    float v = (c < 128) ? si[(size_t)c * N + n0 + m]
                        : sj[(size_t)(c - 128) * N + n0 + m];
    tile[m * CPAD + c] = f2bf(v);
  }
  for (int idx = tid; idx < 16 * 16; idx += blockDim.x) {
    int m = idx >> 4, j = idx & 15;
    *(uint4*)&tile[m * CPAD + 256 + j * 8] =
        *(const uint4*)(xnbf + (size_t)(n0 + m) * 128 + j * 8);
  }
  __syncthreads();
  v8f acc = {};
  acc = wmma_rowtile<CDIM>(W + (size_t)(16 * w) * CDIM, tile, CPAD, lane, acc);
  const int m = n0 + (lane & 15);
  const int ob = 16 * w + (lane >> 4) * 8;
  v8bf pk; float s = 0.f, q = 0.f;
#pragma unroll
  for (int v = 0; v < 8; ++v) { float d = acc[v]; s += d; q += d * d; pk[v] = f2bf(d); }
  *(v8bf*)&T[(size_t)m * ODIM + ob] = pk;
  stats_reduce(s, q, redS, redQ, stats);
}

// ---------- GEMM2 core: tile = relu((T - mean) * rstd) in bf16, then WMMA ----------
template<int CDIM>
__device__ __forceinline__ v8f gemm2_core(const __bf16* __restrict__ T,
                                          const __bf16* __restrict__ W,
                                          const float* __restrict__ stats,
                                          __bf16* tile, int m0) {
  constexpr int CPAD = CDIM + 8;
  const int tid = threadIdx.x, lane = tid & 31, w = tid >> 5;
  const float mean = stats[2], rstd = stats[3];
  for (int idx = tid; idx < 16 * (CDIM / 8); idx += blockDim.x) {
    int m = idx / (CDIM / 8), j = idx % (CDIM / 8);
    v8bf x = *(const v8bf*)(T + (size_t)(m0 + m) * CDIM + j * 8);
    v8bf y;
#pragma unroll
    for (int i = 0; i < 8; ++i) {
      float f = (bf2f(x[i]) - mean) * rstd;
      y[i] = f2bf(fmaxf(f, 0.f));
    }
    *(v8bf*)&tile[m * CPAD + j * 8] = y;
  }
  __syncthreads();
  v8f acc = {};
  return wmma_rowtile<CDIM>(W + (size_t)(16 * w) * CDIM, tile, CPAD, lane, acc);
}

// ---------- opener GEMM2: write Y f32 [128,M] + Ybf [M,128] ----------
__global__ void k_opener2(const __bf16* __restrict__ T, const __bf16* __restrict__ W,
                          const float* __restrict__ stats, float* __restrict__ Y,
                          __bf16* __restrict__ Ybf, int M) {
  constexpr int CDIM = 128;
  __shared__ __bf16 tile[16 * (CDIM + 8)];
  const int lane = threadIdx.x & 31, w = threadIdx.x >> 5;
  const int m0 = blockIdx.x * 16;
  v8f d = gemm2_core<CDIM>(T, W, stats, tile, m0);
  const int m = m0 + (lane & 15);
  const int ob = 16 * w + (lane >> 4) * 8;
  v8bf pk;
#pragma unroll
  for (int v = 0; v < 8; ++v) {
    Y[(size_t)(ob + v) * M + m] = d[v];
    pk[v] = f2bf(d[v]);
  }
  *(v8bf*)&Ybf[(size_t)m * 128 + ob] = pk;
}

// ---------- edge GEMM2: xe += 0.1*xec, refresh xe_bf, scatter-add s_i/s_j ----------
__global__ void k_egemm2(const __bf16* __restrict__ T, const __bf16* __restrict__ W,
                         const float* __restrict__ stats,
                         float* __restrict__ xe, __bf16* __restrict__ xebf,
                         float* __restrict__ si, float* __restrict__ sj,
                         const int* __restrict__ iInd, const int* __restrict__ jInd,
                         int N, int E) {
  constexpr int CDIM = 256;
  __shared__ __bf16 tile[16 * (CDIM + 8)];
  const int lane = threadIdx.x & 31, w = threadIdx.x >> 5;
  const int e0 = blockIdx.x * 16;
  v8f d = gemm2_core<CDIM>(T, W, stats, tile, e0);
  const int e = e0 + (lane & 15);
  const int ob = 16 * w + (lane >> 4) * 8;
  const int ii = iInd[e], jj = jInd[e];
  v8bf pk;
#pragma unroll
  for (int v = 0; v < 8; ++v) {
    float dd = d[v];
    int o = ob + v;
    size_t p = (size_t)o * E + e;
    float xnew = xe[p] + H_STEP * dd;
    xe[p] = xnew;
    pk[v] = f2bf(xnew);
    atomicAdd(&si[(size_t)o * N + ii], dd);
    atomicAdd(&sj[(size_t)o * N + jj], dd);
  }
  *(v8bf*)&xebf[(size_t)e * 128 + ob] = pk;
}

// ---------- node GEMM2: xn += 0.1*xnc, refresh xn_bf ----------
__global__ void k_ngemm2(const __bf16* __restrict__ T, const __bf16* __restrict__ W,
                         const float* __restrict__ stats,
                         float* __restrict__ xn, __bf16* __restrict__ xnbf, int N) {
  constexpr int CDIM = 256;
  __shared__ __bf16 tile[16 * (CDIM + 8)];
  const int lane = threadIdx.x & 31, w = threadIdx.x >> 5;
  const int n0 = blockIdx.x * 16;
  v8f d = gemm2_core<CDIM>(T, W, stats, tile, n0);
  const int n = n0 + (lane & 15);
  const int ob = 16 * w + (lane >> 4) * 8;
  v8bf pk;
#pragma unroll
  for (int v = 0; v < 8; ++v) {
    size_t p = (size_t)(ob + v) * N + n;
    float xnew = xn[p] + H_STEP * d[v];
    xn[p] = xnew;
    pk[v] = f2bf(xnew);
  }
  *(v8bf*)&xnbf[(size_t)n * 128 + ob] = pk;
}

// ---------- output GEMM: [64,128] @ xn ----------
__global__ void k_outgemm(const __bf16* __restrict__ xnbf, const __bf16* __restrict__ W,
                          float* __restrict__ out, int N) {
  constexpr int CDIM = 128, CPAD = CDIM + 8;
  __shared__ __bf16 tile[16 * CPAD];
  const int tid = threadIdx.x, lane = tid & 31, w = tid >> 5;
  const int n0 = blockIdx.x * 16;
  for (int idx = tid; idx < 16 * 16; idx += blockDim.x) {
    int m = idx >> 4, j = idx & 15;
    *(uint4*)&tile[m * CPAD + j * 8] =
        *(const uint4*)(xnbf + (size_t)(n0 + m) * 128 + j * 8);
  }
  __syncthreads();
  v8f acc = {};
  acc = wmma_rowtile<CDIM>(W + (size_t)(16 * w) * CDIM, tile, CPAD, lane, acc);
  const int n = n0 + (lane & 15);
  const int ob = 16 * w + (lane >> 4) * 8;
#pragma unroll
  for (int v = 0; v < 8; ++v) out[(size_t)(ob + v) * N + n] = acc[v];
}

extern "C" void kernel_launch(void* const* d_in, const int* in_sizes, int n_in,
                              void* d_out, int out_size, void* d_ws, size_t ws_size,
                              hipStream_t stream) {
  (void)n_in; (void)out_size; (void)ws_size;
  const float* xn_in = (const float*)d_in[0];
  const float* xe_in = (const float*)d_in[1];
  const int*   iInd  = (const int*)d_in[2];
  const int*   jInd  = (const int*)d_in[3];
  const float* K1N   = (const float*)d_in[4];
  const float* K2N   = (const float*)d_in[5];
  const float* K1E   = (const float*)d_in[6];
  const float* K2E   = (const float*)d_in[7];
  const float* KOUT  = (const float*)d_in[8];
  const float* KE1   = (const float*)d_in[9];
  const float* KE2   = (const float*)d_in[10];
  const float* KN1   = (const float*)d_in[11];
  const float* KN2   = (const float*)d_in[12];

  const int N = in_sizes[0] / 32;   // 50000
  const int E = in_sizes[1] / 32;   // 500000
  const int NL = 4;

  char* p = (char*)d_ws;
  auto carve = [&](size_t bytes) -> char* {
    char* r = p; p += (bytes + 255) & ~(size_t)255; return r;
  };
  float*  stats = (float*)carve(16 * 4 * sizeof(float));
  __bf16* W1N   = (__bf16*)carve((size_t)128 * 32 * 2);
  __bf16* W2N   = (__bf16*)carve((size_t)128 * 128 * 2);
  __bf16* W1E   = (__bf16*)carve((size_t)128 * 32 * 2);
  __bf16* W2E   = (__bf16*)carve((size_t)128 * 128 * 2);
  __bf16* WOUT  = (__bf16*)carve((size_t)64 * 128 * 2);
  __bf16* KE1F  = (__bf16*)carve((size_t)NL * 256 * 256 * 2);
  __bf16* KE2B  = (__bf16*)carve((size_t)NL * 128 * 256 * 2);
  __bf16* KN1F  = (__bf16*)carve((size_t)NL * 256 * 384 * 2);
  __bf16* KN2B  = (__bf16*)carve((size_t)NL * 128 * 256 * 2);
  float*  xn    = (float*)carve((size_t)128 * N * 4);
  float*  s_i   = (float*)carve((size_t)128 * N * 4);
  float*  s_j   = (float*)carve((size_t)128 * N * 4);
  __bf16* xnbf  = (__bf16*)carve((size_t)N * 128 * 2);
  __bf16* tnbf  = (__bf16*)carve((size_t)N * 256 * 2);
  __bf16* xebf  = (__bf16*)carve((size_t)E * 128 * 2);
  __bf16* tebf  = (__bf16*)carve((size_t)E * 256 * 2);

  float* out_xn = (float*)d_out;
  float* xe     = out_xn + (size_t)64 * N;   // xe lives in d_out

  auto cvt = [&](__bf16* d, const float* s, int n) {
    k_cvt<<<(n + 255) / 256, 256, 0, stream>>>(d, s, n);
  };
  cvt(W1N, K1N, 128 * 32); cvt(W2N, K2N, 128 * 128);
  cvt(W1E, K1E, 128 * 32); cvt(W2E, K2E, 128 * 128);
  cvt(WOUT, KOUT, 64 * 128);
  for (int l = 0; l < NL; ++l) {
    k_fold_e<<<(256 * 256 + 255) / 256, 256, 0, stream>>>(
        KE1F + (size_t)l * 256 * 256, KE1 + (size_t)l * 256 * 384);
    k_fold_n<<<(256 * 384 + 255) / 256, 256, 0, stream>>>(
        KN1F + (size_t)l * 256 * 384, KN1 + (size_t)l * 256 * 384);
    cvt(KE2B + (size_t)l * 128 * 256, KE2 + (size_t)l * 128 * 256, 128 * 256);
    cvt(KN2B + (size_t)l * 128 * 256, KN2 + (size_t)l * 128 * 256, 128 * 256);
  }
  hipMemsetAsync(stats, 0, 16 * 4 * sizeof(float), stream);

  // openers
  k_gemm1_f32<32, 128><<<N / 16, 256, 0, stream>>>(xn_in, W1N, tnbf, stats + 0, N);
  k_finalize<<<1, 32, 0, stream>>>(stats + 0, (float)(1.0 / (128.0 * (double)N)));
  k_opener2<<<N / 16, 256, 0, stream>>>(tnbf, W2N, stats + 0, xn, xnbf, N);

  k_gemm1_f32<32, 128><<<E / 16, 256, 0, stream>>>(xe_in, W1E, tebf, stats + 4, E);
  k_finalize<<<1, 32, 0, stream>>>(stats + 4, (float)(1.0 / (128.0 * (double)E)));
  k_opener2<<<E / 16, 256, 0, stream>>>(tebf, W2E, stats + 4, xe, xebf, E);

  for (int l = 0; l < NL; ++l) {
    hipMemsetAsync(s_i, 0, (size_t)128 * N * 4, stream);
    hipMemsetAsync(s_j, 0, (size_t)128 * N * 4, stream);
    float* se = stats + (2 + 2 * l) * 4;
    float* sn = stats + (3 + 2 * l) * 4;
    k_egemm1<<<E / 16, 512, 0, stream>>>(xnbf, xebf, iInd,
                                         KE1F + (size_t)l * 256 * 256, tebf, se, E);
    k_finalize<<<1, 32, 0, stream>>>(se, (float)(1.0 / (256.0 * (double)E)));
    k_egemm2<<<E / 16, 256, 0, stream>>>(tebf, KE2B + (size_t)l * 128 * 256, se,
                                         xe, xebf, s_i, s_j, iInd, jInd, N, E);
    k_ngemm1<<<N / 16, 512, 0, stream>>>(s_i, s_j, xnbf,
                                         KN1F + (size_t)l * 256 * 384, tnbf, sn, N);
    k_finalize<<<1, 32, 0, stream>>>(sn, (float)(1.0 / (256.0 * (double)N)));
    k_ngemm2<<<N / 16, 256, 0, stream>>>(tnbf, KN2B + (size_t)l * 128 * 256, sn,
                                         xn, xnbf, N);
  }
  k_outgemm<<<N / 16, 128, 0, stream>>>(xnbf, WOUT, out_xn, N);
}
